// NSF_AR_81690277969984
// MI455X (gfx1250) — matrix operand order repaired and checked by reading, not compiled, not run on hardware
//
#include <hip/hip_runtime.h>
#include <hip/hip_bf16.h>

typedef __attribute__((ext_vector_type(16))) _Float16 v16h;
typedef __attribute__((ext_vector_type(8)))  _Float16 v8h;
typedef __attribute__((ext_vector_type(8)))  float    v8f;

#define DIM      64
#define FEATDIM  128
#define OUT_DIM  14
#define WIDTH    1536
#define BATCH    32768
#define K1       192            // FEATDIM + DIM
#define NOUT     896            // DIM * OUT_DIM
#define MT       32             // batch rows per block
#define NT12     (WIDTH/16)     // 96 n-tiles for layers 1,2
#define KS1      (K1/32)        // 6 k-steps layer 1
#define KS2      (WIDTH/32)     // 48 k-steps layers 2,3
#define NT3      (NOUT/16)      // 56 n-tiles layer 3
#define TILE_ELEMS 512          // 16(N) x 32(K) f16 per WMMA B tile

#define W1_TILES (NT12*KS1)                       // 576
#define W2_TILES (NT12*KS2)                       // 4608
#define W3_TILES (NT3*KS2)                        // 2688
#define W2_OFF   ((long)W1_TILES*TILE_ELEMS)
#define W3_OFF   ((long)(W1_TILES+W2_TILES)*TILE_ELEMS)
#define TOTAL_TILES (W1_TILES+W2_TILES+W3_TILES)  // 7872

// LDS layout (dynamic): [obuf f32 32x896 | h1 f16 32x1536 | h2 f16 32x1536 | stats]
#define OBUF_BYTES (MT*NOUT*4)     // 114688
#define H_BYTES    (MT*WIDTH*2)    // 98304
#define SMEM_BYTES (OBUF_BYTES + 2*H_BYTES + 1024)  // 312320 <= 320 KB WGP LDS

#define WMMA_F16(a, b, c) \
  __builtin_amdgcn_wmma_f32_16x16x32_f16(false, (a), false, (b), (short)0, (c), false, false)

// ---------------------------------------------------------------------------
// Prep: masked f32 weights -> f16, pre-swizzled into WMMA B-fragment order.
// Tile (nt,kt): lane l's 16 contiguous f16 hold W[n = nt*16 + l%16]
//                                            [k = kt*32 + (l/16)*8 + (e%8) + (e/8)*16]
// ---------------------------------------------------------------------------
__global__ void nsf_prep_weights(const float* __restrict__ w1, const float* __restrict__ m1,
                                 const float* __restrict__ w2, const float* __restrict__ m2,
                                 const float* __restrict__ w3, const float* __restrict__ m3,
                                 _Float16* __restrict__ dst) {
  const long total = (long)TOTAL_TILES * TILE_ELEMS;
  for (long i = (long)blockIdx.x * blockDim.x + threadIdx.x; i < total;
       i += (long)gridDim.x * blockDim.x) {
    long tile = i >> 9;
    int  pos  = (int)(i & 511);
    int  lane = pos >> 4, e = pos & 15;
    const float *W, *M;
    int kdim, ksteps; long tloc;
    if (tile < W1_TILES)                { W = w1; M = m1; kdim = K1;    ksteps = KS1; tloc = tile; }
    else if (tile < W1_TILES+W2_TILES)  { W = w2; M = m2; kdim = WIDTH; ksteps = KS2; tloc = tile - W1_TILES; }
    else                                { W = w3; M = m3; kdim = WIDTH; ksteps = KS2; tloc = tile - (W1_TILES+W2_TILES); }
    int nt = (int)(tloc / ksteps), kt = (int)(tloc % ksteps);
    int n = nt*16 + (lane & 15);
    int k = kt*32 + ((lane >> 4) << 3) + (e & 7) + ((e >> 3) << 4);
    long src = (long)n * kdim + k;
    dst[i] = (_Float16)(W[src] * M[src]);
  }
}

// A-fragment (16x32 f16, ISA layout) from an LDS activation buffer.
__device__ __forceinline__ v16h load_a(const _Float16* h, int stride, int mt, int kt, int lane) {
  const int m  = mt*16 + (lane & 15);
  const int k0 = kt*32 + ((lane >> 4) << 3);
  const v8h lo = *(const v8h*)(h + m*stride + k0);
  const v8h hi = *(const v8h*)(h + m*stride + k0 + 16);
  v16h a;
#pragma unroll
  for (int i = 0; i < 8; ++i) { a[i] = lo[i]; a[i+8] = hi[i]; }
  return a;
}

// GEMM (MT x ndim): two independent WMMA accumulator chains per wave iteration
// (tiles t and t+8 share mt and the A fragment). LN stats accumulate in
// registers across the whole GEMM; one cross-lane reduction at the end.
__device__ __forceinline__ void gemm_act_stage(
    const _Float16* __restrict__ hsrc, int kdim, int ksteps,
    const _Float16* __restrict__ wtiles, int ntiles,
    const float* __restrict__ bias,
    _Float16* __restrict__ hdst, float* ssum, float* ssq, int wave, int lane) {
  const int ndim  = ntiles * 16;
  const int tiles = 2 * ntiles;          // 2 m-tiles (MT=32)
  const int mt    = wave & 1;            // tile stride 8/16 keeps parity fixed
  float sacc[8], qacc[8];
#pragma unroll
  for (int r = 0; r < 8; ++r) { sacc[r] = 0.f; qacc[r] = 0.f; }

  for (int t = wave; t < tiles; t += 16) {
    const int nt0 = t >> 1, nt1 = nt0 + 4;
    v8f acc0 = {}, acc1 = {};
    const _Float16* wb0 = wtiles + (long)(nt0*ksteps)*TILE_ELEMS + lane*16;
    const _Float16* wb1 = wtiles + (long)(nt1*ksteps)*TILE_ELEMS + lane*16;
#pragma unroll 4
    for (int kt = 0; kt < ksteps; ++kt) {
      v16h a  = load_a(hsrc, kdim, mt, kt, lane);
      v16h b0 = *(const v16h*)(wb0 + kt*TILE_ELEMS);
      v16h b1 = *(const v16h*)(wb1 + kt*TILE_ELEMS);
      acc0 = WMMA_F16(a, b0, acc0);
      acc1 = WMMA_F16(a, b1, acc1);
    }
    const int   n0  = nt0*16 + (lane & 15), n1 = nt1*16 + (lane & 15);
    const float bn0 = bias[n0],             bn1 = bias[n1];
#pragma unroll
    for (int r = 0; r < 8; ++r) {
      const int m = mt*16 + r + ((lane >> 4) << 3);
      float c0 = acc0[r] + bn0;
      c0 = (c0 > 0.f) ? c0 : 0.2f * c0;                // leaky_relu 0.2
      hdst[m*ndim + n0] = (_Float16)c0;
      float c1 = acc1[r] + bn1;
      c1 = (c1 > 0.f) ? c1 : 0.2f * c1;
      hdst[m*ndim + n1] = (_Float16)c1;
      sacc[r] += c0 + c1;
      qacc[r] += c0*c0 + c1*c1;
    }
  }

  // one reduction per row at the end (16-lane halves -> one atomic each)
#pragma unroll
  for (int r = 0; r < 8; ++r) {
    float s = sacc[r], q = qacc[r];
#pragma unroll
    for (int off = 1; off < 16; off <<= 1) {
      s += __shfl_xor(s, off, 32);
      q += __shfl_xor(q, off, 32);
    }
    if ((lane & 15) == 0) {
      const int m = mt*16 + r + ((lane >> 4) << 3);
      atomicAdd(&ssum[m], s);
      atomicAdd(&ssq[m], q);
    }
  }
}

__device__ __forceinline__ void ln_apply(_Float16* h, int ndim,
                                         const float* __restrict__ g, const float* __restrict__ b,
                                         const float* smean, const float* srstd, int tid) {
  for (int i = tid; i < MT*ndim; i += 256) {
    int m = i / ndim, n = i - m*ndim;
    float v = (float)h[i];
    v = (v - smean[m]) * srstd[m] * g[n] + b[n];
    h[i] = (_Float16)v;
  }
}

// ---------------------------------------------------------------------------
// Fully fused: GEMM1 -> LN -> GEMM2 -> LN -> GEMM3 -> RQ-spline + logdet
// ---------------------------------------------------------------------------
__global__ __launch_bounds__(256) void nsf_fused(
    const float* __restrict__ x,  const float* __restrict__ feat,
    const float* __restrict__ b1, const float* __restrict__ g1, const float* __restrict__ be1,
    const float* __restrict__ b2, const float* __restrict__ g2, const float* __restrict__ be2,
    const float* __restrict__ b3,
    const _Float16* __restrict__ w1s, const _Float16* __restrict__ w2s,
    const _Float16* __restrict__ w3s,
    float* __restrict__ zout, float* __restrict__ ladout) {
  extern __shared__ char smem[];
  float*    obuf = (float*)smem;                       // 32x896 f32 (phase 3)
  _Float16* hA   = (_Float16*)smem;                    // 32x192 f16 (phase 1, overlaps obuf)
  _Float16* h1   = (_Float16*)(smem + OBUF_BYTES);
  _Float16* h2   = (_Float16*)(smem + OBUF_BYTES + H_BYTES);
  float*    ssum = (float*)(smem + OBUF_BYTES + 2*H_BYTES);
  float*    ssq  = ssum + 32;
  float*    smean= ssum + 64;
  float*    srstd= ssum + 96;
  float*    slad = ssum + 128;

  const int tid  = threadIdx.x;
  const int lane = tid & 31;
  const int wave = tid >> 5;
  const int row0 = blockIdx.x * MT;

  // ---- stage concat(feat, x) -> LDS f16; zero stats
  for (int i = tid; i < MT*K1; i += 256) {
    int m = i / K1, c = i - m*K1;
    float v = (c < FEATDIM) ? feat[(long)(row0+m)*FEATDIM + c]
                            : x[(long)(row0+m)*DIM + (c - FEATDIM)];
    hA[m*K1 + c] = (_Float16)v;
  }
  if (tid < 32) { ssum[tid] = 0.f; ssq[tid] = 0.f; }
  __syncthreads();

  // ---- layer 1
  gemm_act_stage(hA, K1, KS1, w1s, NT12, b1, h1, ssum, ssq, wave, lane);
  __syncthreads();
  if (tid < 32) {
    float mean = ssum[tid] * (1.f/WIDTH);
    float var  = ssq[tid]  * (1.f/WIDTH) - mean*mean;
    smean[tid] = mean; srstd[tid] = rsqrtf(var + 1e-5f);
    ssum[tid] = 0.f; ssq[tid] = 0.f;
  }
  __syncthreads();
  ln_apply(h1, WIDTH, g1, be1, smean, srstd, tid);
  __syncthreads();

  // ---- layer 2
  gemm_act_stage(h1, WIDTH, KS2, w2s, NT12, b2, h2, ssum, ssq, wave, lane);
  __syncthreads();
  if (tid < 32) {
    float mean = ssum[tid] * (1.f/WIDTH);
    float var  = ssq[tid]  * (1.f/WIDTH) - mean*mean;
    smean[tid] = mean; srstd[tid] = rsqrtf(var + 1e-5f);
    slad[tid] = 0.f;
  }
  __syncthreads();
  ln_apply(h2, WIDTH, g2, be2, smean, srstd, tid);
  __syncthreads();

  // ---- layer 3: o = h2 @ W3^T + b3 -> obuf (f32, overlaps retired hA)
  {
    const int mt = wave & 1;
    for (int t = wave; t < 2*NT3; t += 16) {
      const int nt0 = t >> 1, nt1 = nt0 + 4;
      v8f acc0 = {}, acc1 = {};
      const _Float16* wb0 = w3s + (long)(nt0*KS2)*TILE_ELEMS + lane*16;
      const _Float16* wb1 = w3s + (long)(nt1*KS2)*TILE_ELEMS + lane*16;
#pragma unroll 4
      for (int kt = 0; kt < KS2; ++kt) {
        v16h a  = load_a(h2, WIDTH, mt, kt, lane);
        v16h b0 = *(const v16h*)(wb0 + kt*TILE_ELEMS);
        v16h b1 = *(const v16h*)(wb1 + kt*TILE_ELEMS);
        acc0 = WMMA_F16(a, b0, acc0);
        acc1 = WMMA_F16(a, b1, acc1);
      }
      const int   n0  = nt0*16 + (lane & 15), n1 = nt1*16 + (lane & 15);
      const float bn0 = b3[n0],               bn1 = b3[n1];
#pragma unroll
      for (int r = 0; r < 8; ++r) {
        const int m = mt*16 + r + ((lane >> 4) << 3);
        obuf[m*NOUT + n0] = acc0[r] + bn0;
        obuf[m*NOUT + n1] = acc1[r] + bn1;
      }
    }
  }
  __syncthreads();

  // ---- rational-quadratic spline per (row, dim)
  for (int i = tid; i < MT*DIM; i += 256) {
    const int m = i >> 6, d = i & 63;
    const float* o = obuf + m*NOUT + d*OUT_DIM;   // [W(5) | H(5) | D(4)]
    // widths softmax -> cumwidths
    float mw = o[0];
#pragma unroll
    for (int k = 1; k < 5; ++k) mw = fmaxf(mw, o[k]);
    float ew[5], sw = 0.f;
#pragma unroll
    for (int k = 0; k < 5; ++k) { ew[k] = __expf(o[k]-mw); sw += ew[k]; }
    float cw[6]; cw[0] = 0.f;
    { float c = 0.f, sc = (1.f - 5.f*0.001f)/sw;
#pragma unroll
      for (int k = 0; k < 5; ++k) { c += 0.001f + sc*ew[k]; cw[k+1] = c; } }
    cw[5] = 1.0f;
    // heights softmax -> cumheights
    float mh = o[5];
#pragma unroll
    for (int k = 1; k < 5; ++k) mh = fmaxf(mh, o[5+k]);
    float eh[5], sh = 0.f;
#pragma unroll
    for (int k = 0; k < 5; ++k) { eh[k] = __expf(o[5+k]-mh); sh += eh[k]; }
    float ch[6]; ch[0] = 0.f;
    { float c = 0.f, sc = (1.f - 5.f*0.001f)/sh;
#pragma unroll
      for (int k = 0; k < 5; ++k) { c += 0.001f + sc*eh[k]; ch[k+1] = c; } }
    ch[5] = 1.0f;
    // derivatives: pads are exactly MIN_D + softplus(CONST) = 1.0
    float dv[6]; dv[0] = 1.0f; dv[5] = 1.0f;
#pragma unroll
    for (int k = 0; k < 4; ++k) {
      float u = o[10+k];
      float sp = (u > 20.f) ? u : log1pf(__expf(u));
      dv[k+1] = 0.001f + sp;
    }
    const float xx = x[(long)(row0+m)*DIM + d];
    const float s  = 1.f/(1.f + __expf(-xx));
    const float logds = -log1pf(__expf(-xx)) - log1pf(__expf(xx));
    int idx = 0;
#pragma unroll
    for (int k = 1; k < 5; ++k) idx = (s >= cw[k]) ? k : idx;
    const float in_cw = cw[idx], in_w = cw[idx+1]-cw[idx];
    const float in_ch = ch[idx], in_h = ch[idx+1]-ch[idx];
    const float delta = in_h / in_w;
    const float d0 = dv[idx], d1 = dv[idx+1];
    const float theta = (s - in_cw) / in_w;
    const float omt = 1.f - theta, t1mt = theta*omt;
    const float num = in_h * (delta*theta*theta + d0*t1mt);
    const float den = delta + (d0 + d1 - 2.f*delta)*t1mt;
    float outv = in_ch + num/den;
    const float dnum = delta*delta*(d1*theta*theta + 2.f*delta*t1mt + d0*omt*omt);
    float lad = logds + __logf(dnum) - 2.f*__logf(den);
    outv = outv*0.999998f + 1e-6f;
    const float lo_ = __logf(outv), l1_ = __logf(1.f - outv);
    zout[(long)(row0+m)*DIM + d] = lo_ - l1_;
    lad += -lo_ - l1_;
    atomicAdd(&slad[m], lad);
  }
  __syncthreads();
  if (tid < 32) ladout[row0 + tid] = slad[tid];
}

extern "C" void kernel_launch(void* const* d_in, const int* in_sizes, int n_in,
                              void* d_out, int out_size, void* d_ws, size_t ws_size,
                              hipStream_t stream) {
  const float* x   = (const float*)d_in[0];
  const float* feat= (const float*)d_in[1];
  const float* w1  = (const float*)d_in[2];
  const float* b1  = (const float*)d_in[3];
  const float* g1  = (const float*)d_in[4];
  const float* be1 = (const float*)d_in[5];
  const float* w2  = (const float*)d_in[6];
  const float* b2  = (const float*)d_in[7];
  const float* g2  = (const float*)d_in[8];
  const float* be2 = (const float*)d_in[9];
  const float* w3  = (const float*)d_in[10];
  const float* b3  = (const float*)d_in[11];
  const float* m1  = (const float*)d_in[12];
  const float* m2  = (const float*)d_in[13];
  const float* m3  = (const float*)d_in[14];

  _Float16* wtiles = (_Float16*)d_ws;      // ~8.06 MB of swizzled f16 weights
  float* zout   = (float*)d_out;
  float* ladout = zout + (long)BATCH*DIM;

  nsf_prep_weights<<<2048, 256, 0, stream>>>(w1, m1, w2, m2, w3, m3, wtiles);

  hipFuncSetAttribute(reinterpret_cast<const void*>(nsf_fused),
                      hipFuncAttributeMaxDynamicSharedMemorySize, SMEM_BYTES);
  nsf_fused<<<BATCH/MT, 256, SMEM_BYTES, stream>>>(
      x, feat, b1, g1, be1, b2, g2, be2, b3,
      wtiles, wtiles + W2_OFF, wtiles + W3_OFF, zout, ladout);
}
